// Network_fc_bin_W_N_31945966748049
// MI455X (gfx1250) — 16-bit and 8-bit matrix operand order repaired and checked by reading, compiled, not run on hardware
//
#include <hip/hip_runtime.h>
#include <hip/hip_bf16.h>

typedef __attribute__((ext_vector_type(8)))  int    v8i;
typedef __attribute__((ext_vector_type(16))) int    v16i;
typedef __attribute__((ext_vector_type(8)))  float  v8f;
typedef __attribute__((ext_vector_type(16))) __bf16 v16bf;

#define B_ROWS 16384
#define MT     1024
#define HID    512
#define KT8    4
#define DIN    784
#define KTD    25

#define FP8_P1 0x38u
#define FP8_M1 0xB8u

#define OFF_BINA1 0u
#define OFF_BINA2 2097152u
#define OFF_BINA0 4194304u
#define OFF_W1T   4718592u
#define OFF_W1    4784128u
#define OFF_W0T   4849664u
#define OFF_W0    4851712u
#define OFF_W2P   4868096u
#define OFF_C2    5072896u

__global__ void __launch_bounds__(256)
pack_b_fp8(const float* __restrict__ W, int rows, int cols, int trans,
           int KT, unsigned int* __restrict__ out, int total) {
  int t = blockIdx.x * blockDim.x + threadIdx.x;
  if (t >= total) return;
  int j    = t & 15;
  int lane = (t >> 4) & 31;
  int kt   = (t >> 9) % KT;
  int nt   = (t >> 9) / KT;
  int jj   = j & 7;
  int n = nt * 16 + (lane & 15);
  int kbase = kt * 128 + (j >> 3) * 64 + (jj & 3) * 4 + (jj >> 2) * 32 + ((lane >> 4) & 1) * 16;
  unsigned int dw = 0;
  #pragma unroll
  for (int i = 0; i < 4; ++i) {
    int k = kbase + i;
    unsigned int bv = 0;
    if (trans) { if (n < rows && k < cols) bv = (W[n * cols + k] > 0.f) ? FP8_P1 : FP8_M1; }
    else       { if (k < rows && n < cols) bv = (W[k * cols + n] > 0.f) ? FP8_P1 : FP8_M1; }
    dw |= bv << (8 * i);
  }
  out[t] = dw;
}

__global__ void __launch_bounds__(256)
pack_b_bf16_t(const float* __restrict__ W, int rows, int cols,
              int KT, unsigned int* __restrict__ out, int total) {
  int t = blockIdx.x * blockDim.x + threadIdx.x;
  if (t >= total) return;
  int j    = t & 7;
  int lane = (t >> 3) & 31;
  int kt   = (t >> 8) % KT;
  int nt   = (t >> 8) / KT;
  int n = nt * 16 + (lane & 15);
  unsigned int dw = 0;
  #pragma unroll
  for (int e = 0; e < 2; ++e) {
    int k = kt * 32 + j * 2 + ((lane >> 4) & 1) * 16 + e;
    unsigned int h = 0;
    if (n < rows && k < cols) h = (W[n * cols + k] > 0.f) ? 0x3F80u : 0xBF80u;
    dw |= h << (16 * e);
  }
  out[t] = dw;
}

__global__ void __launch_bounds__(256)
binarize_pack(const float* __restrict__ s, int cols, int KT,
              unsigned int* __restrict__ out, int total) {
  int t = blockIdx.x * blockDim.x + threadIdx.x;
  if (t >= total) return;
  int j    = t & 15;
  int lane = (t >> 4) & 31;
  int kt   = (t >> 9) % KT;
  int mt   = (t >> 9) / KT;
  int jj   = j & 7;
  int m = mt * 16 + (lane & 15);
  int kbase = kt * 128 + (j >> 3) * 64 + (jj & 3) * 4 + (jj >> 2) * 32 + ((lane >> 4) & 1) * 16;
  unsigned int dw = 0;
  #pragma unroll
  for (int i = 0; i < 4; ++i) {
    int k = kbase + i;
    unsigned int b = (k < cols && s[m * cols + k] >= 0.5f) ? FP8_P1 : 0u;
    dw |= b << (8 * i);
  }
  out[t] = dw;
}

__global__ void __launch_bounds__(256)
copy_f32(const float* __restrict__ src, float* __restrict__ dst, int n) {
  int t = blockIdx.x * blockDim.x + threadIdx.x;
  if (t < n) dst[t] = src[t];
}

__global__ void __launch_bounds__(256)
c2_gemm(const float* __restrict__ data, const unsigned int* __restrict__ w2p,
        const float* __restrict__ W2, const float* __restrict__ b2,
        float* __restrict__ c2) {
  int tid  = threadIdx.x;
  int lane = tid & 31;
  int wv   = (blockIdx.x * 256 + tid) >> 5;
  int mt   = wv >> 3;
  int ng   = wv & 7;
  int hi   = (lane >> 4) & 1;
  int m    = mt * 16 + (lane & 15);

  v8f acc[4] = {};
  for (int kt = 0; kt < KTD; ++kt) {
    v16bf ahi, alo;
    #pragma unroll
    for (int v = 0; v < 8; ++v) {
      int kb = kt * 32 + v * 2 + hi * 16;
      float x0 = (kb     < DIN) ? data[m * DIN + kb]     : 0.f;
      float x1 = (kb + 1 < DIN) ? data[m * DIN + kb + 1] : 0.f;
      __bf16 h0 = (__bf16)x0, h1 = (__bf16)x1;
      ahi[2 * v]     = h0;
      ahi[2 * v + 1] = h1;
      alo[2 * v]     = (__bf16)(x0 - (float)h0);
      alo[2 * v + 1] = (__bf16)(x1 - (float)h1);
    }
    #pragma unroll
    for (int t = 0; t < 4; ++t) {
      v16bf b = *(const v16bf*)(w2p + ((((size_t)(ng * 4 + t) * KTD + kt) * 32 + lane) * 8));
      acc[t] = __builtin_amdgcn_wmma_f32_16x16x32_bf16(false, ahi, false, b, (short)0, acc[t], false, false);
      acc[t] = __builtin_amdgcn_wmma_f32_16x16x32_bf16(false, alo, false, b, (short)0, acc[t], false, false);
    }
  }
  float off2 = fabsf(W2[0]);
  #pragma unroll
  for (int t = 0; t < 4; ++t)
    #pragma unroll
    for (int v = 0; v < 8; ++v) {
      int mm = mt * 16 + v + 8 * hi;
      int n  = (ng * 4 + t) * 16 + (lane & 15);
      c2[mm * HID + n] = off2 * acc[t][v] + b2[n];
    }
}

__global__ void __launch_bounds__(256)
step_s2(const unsigned int* __restrict__ binA1, const unsigned int* __restrict__ w1p,
        const float* __restrict__ c2, const float* __restrict__ W1,
        float* __restrict__ s2) {
  int tid = threadIdx.x, lane = tid & 31;
  int wv = (blockIdx.x * 256 + tid) >> 5;
  int mt = wv >> 3, ng = wv & 7;
  int hi = (lane >> 4) & 1;

  v8f acc[4] = {};
  for (int kt = 0; kt < KT8; ++kt) {
    v16i a = *(const v16i*)(binA1 + (((size_t)mt * KT8 + kt) * 32 + lane) * 16);
    #pragma unroll
    for (int t = 0; t < 4; ++t) {
      v16i b = *(const v16i*)(w1p + ((((size_t)(ng * 4 + t) * KT8 + kt) * 32 + lane) * 16));
      acc[t] = __builtin_amdgcn_wmma_f32_16x16x128_fp8_fp8(a, b, (short)0, acc[t], false, false);
    }
  }
  float off1 = fabsf(W1[0]);
  #pragma unroll
  for (int t = 0; t < 4; ++t)
    #pragma unroll
    for (int v = 0; v < 8; ++v) {
      int mm = mt * 16 + v + 8 * hi;
      int n  = (ng * 4 + t) * 16 + (lane & 15);
      int idx = mm * HID + n;
      float val = c2[idx] + off1 * acc[t][v];
      float so  = s2[idx];
      float x   = so - 0.5f;
      float rp  = (x >= -0.5f && x <= 0.5f) ? 1.f : 0.f;
      float ns  = 0.5f * so + 0.5f * (rp * val);
      s2[idx] = fminf(fmaxf(ns, 0.f), 1.f);
    }
}

__global__ void __launch_bounds__(256)
step_s1(const unsigned int* __restrict__ binA2, const unsigned int* __restrict__ w1tp,
        const unsigned int* __restrict__ binA0, const unsigned int* __restrict__ w0p,
        const float* __restrict__ W1, const float* __restrict__ W0,
        const float* __restrict__ b1, float* __restrict__ s1) {
  int tid = threadIdx.x, lane = tid & 31;
  int wv = (blockIdx.x * 256 + tid) >> 5;
  int mt = wv >> 3, ng = wv & 7;
  int hi = (lane >> 4) & 1;

  v8f accA[4] = {}, accB[4] = {};
  for (int kt = 0; kt < KT8; ++kt) {
    v16i a = *(const v16i*)(binA2 + (((size_t)mt * KT8 + kt) * 32 + lane) * 16);
    #pragma unroll
    for (int t = 0; t < 4; ++t) {
      v16i b = *(const v16i*)(w1tp + ((((size_t)(ng * 4 + t) * KT8 + kt) * 32 + lane) * 16));
      accA[t] = __builtin_amdgcn_wmma_f32_16x16x128_fp8_fp8(a, b, (short)0, accA[t], false, false);
    }
  }
  {
    v16i a0 = *(const v16i*)(binA0 + (((size_t)mt * 32 + lane) * 16));
    #pragma unroll
    for (int t = 0; t < 4; ++t) {
      v16i b = *(const v16i*)(w0p + ((((size_t)(ng * 4 + t)) * 32 + lane) * 16));
      accB[t] = __builtin_amdgcn_wmma_f32_16x16x128_fp8_fp8(a0, b, (short)0, accB[t], false, false);
    }
  }
  float off1 = fabsf(W1[0]);
  float off0 = fabsf(W0[0]);
  #pragma unroll
  for (int t = 0; t < 4; ++t)
    #pragma unroll
    for (int v = 0; v < 8; ++v) {
      int mm = mt * 16 + v + 8 * hi;
      int n  = (ng * 4 + t) * 16 + (lane & 15);
      int idx = mm * HID + n;
      float val = off1 * accA[t][v] + off0 * accB[t][v] + b1[n];
      float so  = s1[idx];
      float x   = so - 0.5f;
      float rp  = (x >= -0.5f && x <= 0.5f) ? 1.f : 0.f;
      float ns  = 0.5f * so + 0.5f * (rp * val);
      s1[idx] = fminf(fmaxf(ns, 0.f), 1.f);
    }
}

__global__ void __launch_bounds__(256)
step_s0(const unsigned int* __restrict__ binA1, const unsigned int* __restrict__ w0tp,
        const float* __restrict__ W0, const float* __restrict__ b0,
        float* __restrict__ s0) {
  int tid = threadIdx.x, lane = tid & 31;
  int mt = (blockIdx.x * 256 + tid) >> 5;
  int hi = (lane >> 4) & 1;

  v8f acc = {};
  for (int kt = 0; kt < KT8; ++kt) {
    v16i a = *(const v16i*)(binA1 + (((size_t)mt * KT8 + kt) * 32 + lane) * 16);
    v16i b = *(const v16i*)(w0tp + (((size_t)kt * 32 + lane) * 16));
    acc = __builtin_amdgcn_wmma_f32_16x16x128_fp8_fp8(a, b, (short)0, acc, false, false);
  }
  float off0 = fabsf(W0[0]);
  int n = lane & 15;
  if (n < 10) {
    #pragma unroll
    for (int v = 0; v < 8; ++v) {
      int mm = mt * 16 + v + 8 * hi;
      int idx = mm * 10 + n;
      float val = off0 * acc[v] + b0[n];
      float so  = s0[idx];
      float x   = so - 0.5f;
      float rp  = (x >= -0.5f && x <= 0.5f) ? 1.f : 0.f;
      float ns  = 0.5f * so + 0.5f * (rp * val);
      s0[idx] = fminf(fmaxf(ns, 0.f), 1.f);
    }
  }
}

extern "C" void kernel_launch(void* const* d_in, const int* in_sizes, int n_in,
                              void* d_out, int out_size, void* d_ws, size_t ws_size,
                              hipStream_t stream) {
  const float* s0_in = (const float*)d_in[0];
  const float* s1_in = (const float*)d_in[1];
  const float* s2_in = (const float*)d_in[2];
  const float* data  = (const float*)d_in[3];
  const float* W0    = (const float*)d_in[4];
  const float* W1    = (const float*)d_in[5];
  const float* W2    = (const float*)d_in[6];
  const float* b0    = (const float*)d_in[7];
  const float* b1    = (const float*)d_in[8];
  const float* b2    = (const float*)d_in[9];

  float* s0 = (float*)d_out;
  float* s1 = s0 + (size_t)B_ROWS * 10;
  float* s2 = s1 + (size_t)B_ROWS * HID;

  unsigned int* ws = (unsigned int*)d_ws;
  unsigned int* binA1 = ws + OFF_BINA1;
  unsigned int* binA2 = ws + OFF_BINA2;
  unsigned int* binA0 = ws + OFF_BINA0;
  unsigned int* w1tp  = ws + OFF_W1T;
  unsigned int* w1p   = ws + OFF_W1;
  unsigned int* w0tp  = ws + OFF_W0T;
  unsigned int* w0p   = ws + OFF_W0;
  unsigned int* w2p   = ws + OFF_W2P;
  float*        c2    = (float*)(ws + OFF_C2);

  pack_b_fp8<<<(32 * KT8 * 512 + 255) / 256, 256, 0, stream>>>(W1, 512, 512, 1, KT8, w1tp, 32 * KT8 * 512);
  pack_b_fp8<<<(32 * KT8 * 512 + 255) / 256, 256, 0, stream>>>(W1, 512, 512, 0, KT8, w1p,  32 * KT8 * 512);
  pack_b_fp8<<<(1 * KT8 * 512 + 255) / 256, 256, 0, stream>>>(W0, 10, 512, 1, KT8, w0tp, 1 * KT8 * 512);
  pack_b_fp8<<<(32 * 1 * 512 + 255) / 256, 256, 0, stream>>>(W0, 10, 512, 0, 1,   w0p,  32 * 1 * 512);
  pack_b_bf16_t<<<(32 * KTD * 256 + 255) / 256, 256, 0, stream>>>(W2, 512, 784, KTD, w2p, 32 * KTD * 256);

  c2_gemm<<<1024, 256, 0, stream>>>(data, w2p, W2, b2, c2);

  copy_f32<<<(B_ROWS * 10 + 255) / 256, 256, 0, stream>>>(s0_in, s0, B_ROWS * 10);
  copy_f32<<<(B_ROWS * HID + 255) / 256, 256, 0, stream>>>(s1_in, s1, B_ROWS * HID);
  copy_f32<<<(B_ROWS * HID + 255) / 256, 256, 0, stream>>>(s2_in, s2, B_ROWS * HID);

  for (int it = 0; it < 15; ++it) {
    binarize_pack<<<MT * KT8 * 2, 256, 0, stream>>>(s1, HID, KT8, binA1, MT * KT8 * 512);
    binarize_pack<<<MT * KT8 * 2, 256, 0, stream>>>(s2, HID, KT8, binA2, MT * KT8 * 512);
    binarize_pack<<<MT * 2,       256, 0, stream>>>(s0, 10,  1,   binA0, MT * 512);
    step_s1<<<1024, 256, 0, stream>>>(binA2, w1tp, binA0, w0p, W1, W0, b1, s1);
    step_s2<<<1024, 256, 0, stream>>>(binA1, w1p, c2, W1, s2);
    step_s0<<<128,  256, 0, stream>>>(binA1, w0tp, W0, b0, s0);
  }
}
